// GPT_14637248545155
// MI455X (gfx1250) — matrix operand order repaired and checked by reading, not compiled, not run on hardware
//
#include <hip/hip_runtime.h>
#include <math.h>

#define LAYERS 4
#define BATCH  2
#define SEQ    1024
#define CH     768
#define NH     12
#define HD     64
#define VOC    50257
#define MROWS  (BATCH*SEQ)   // 2048

typedef __bf16 bf16_t;
typedef __attribute__((ext_vector_type(4)))  __bf16 v4bf;
typedef __attribute__((ext_vector_type(8)))  __bf16 v8bf;
typedef __attribute__((ext_vector_type(16))) __bf16 v16bf;
typedef __attribute__((ext_vector_type(8)))  float  v8f;
typedef int v4i_t __attribute__((vector_size(16)));

#if defined(__has_builtin)
#if __has_builtin(__builtin_amdgcn_global_load_async_to_lds_b128)
#define HAVE_ASYNC_LDS 1
#endif
#endif
#ifndef HAVE_ASYNC_LDS
#define HAVE_ASYNC_LDS 0
#endif

__device__ __forceinline__ v16bf cat16(v8bf lo, v8bf hi) {
  return __builtin_shufflevector(lo, hi, 0,1,2,3,4,5,6,7,8,9,10,11,12,13,14,15);
}

// 8 contiguous fp32 -> 8 bf16 (with scale), via two 16B loads
__device__ __forceinline__ v8bf cvt8(const float* p, float s) {
  float4 a = *(const float4*)p;
  float4 b = *(const float4*)(p + 4);
  v8bf r;
  r[0]=(__bf16)(a.x*s); r[1]=(__bf16)(a.y*s); r[2]=(__bf16)(a.z*s); r[3]=(__bf16)(a.w*s);
  r[4]=(__bf16)(b.x*s); r[5]=(__bf16)(b.y*s); r[6]=(__bf16)(b.z*s); r[7]=(__bf16)(b.w*s);
  return r;
}

__device__ __forceinline__ v4bf cvt4(float4 a) {
  v4bf r;
  r[0]=(__bf16)a.x; r[1]=(__bf16)a.y; r[2]=(__bf16)a.z; r[3]=(__bf16)a.w;
  return r;
}

// A-matrix fragment (16x32 bf16): lane m = lane%16; lanes<16 hold K {kb..kb+7, kb+16..kb+23}
__device__ __forceinline__ v16bf ldsA16(const bf16_t* base) {
  v8bf lo = *(const v8bf*)(base);
  v8bf hi = *(const v8bf*)(base + 16);
  return cat16(lo, hi);
}
// B-matrix fragment (32x16 bf16): lane n = lane%16; lanes<16 hold K {kb..kb+7, kb+16..kb+23}
__device__ __forceinline__ v16bf ldsB16(const bf16_t* base) {
  v8bf lo = *(const v8bf*)(base);
  v8bf hi = *(const v8bf*)(base + 16);
  return cat16(lo, hi);
}

__device__ __forceinline__ v8f wmma_bf16(v16bf a, v16bf b, v8f c) {
  return __builtin_amdgcn_wmma_f32_16x16x32_bf16(false, a, false, b, (short)0, c, false, false);
}

// async (or sync fallback) 16B global -> LDS copy
__device__ __forceinline__ void cp16_g2l(const bf16_t* g, bf16_t* l) {
#if HAVE_ASYNC_LDS
  __builtin_amdgcn_global_load_async_to_lds_b128(
      (__attribute__((address_space(1))) v4i_t*)(unsigned long long)g,
      (__attribute__((address_space(3))) v4i_t*)l,
      0, 0);
#else
  *(uint4*)l = *(const uint4*)g;
#endif
}

template <int N>
__device__ __forceinline__ void wait_async() {
#if __has_builtin(__builtin_amdgcn_s_wait_asynccnt)
  __builtin_amdgcn_s_wait_asynccnt(N);
#else
  asm volatile("s_wait_asynccnt %0" :: "i"(N) : "memory");
#endif
}

// ---------------------------------------------------------------------------
// fp32 [K,N] -> bf16 [Npad,K] transposed convert; rows [N,Npad) zero-filled
// ---------------------------------------------------------------------------
__global__ __launch_bounds__(256)
void transpose_cvt(const float* __restrict__ W, bf16_t* __restrict__ WT,
                   int K, int N, int Npad) {
  __shared__ float tile[32][33];
  int n0 = blockIdx.x * 32, k0 = blockIdx.y * 32;
  int tx = threadIdx.x, ty = threadIdx.y;   // block (32,8)
  #pragma unroll
  for (int i = 0; i < 4; ++i) {
    int k = k0 + ty + i*8, n = n0 + tx;
    tile[ty + i*8][tx] = (k < K && n < N) ? W[(size_t)k*N + n] : 0.f;
  }
  __syncthreads();
  #pragma unroll
  for (int i = 0; i < 4; ++i) {
    int n = n0 + ty + i*8, k = k0 + tx;
    if (n < Npad && k < K) WT[(size_t)n*K + k] = (bf16_t)tile[tx][ty + i*8];
  }
}

// ---------------------------------------------------------------------------
// x = wte[idx] + wpe
// ---------------------------------------------------------------------------
__global__ void embed_kernel(const int* __restrict__ idx, const float* __restrict__ wte,
                             const float* __restrict__ wpe, float* __restrict__ x) {
  int i = blockIdx.x * 256 + threadIdx.x;
  if (i >= MROWS * CH) return;
  int c = i % CH;
  int rt = i / CH;          // b*SEQ + t
  int t = rt % SEQ;
  int tok = idx[rt];
  x[i] = wte[(size_t)tok*CH + c] + wpe[(size_t)t*CH + c];
}

// ---------------------------------------------------------------------------
// LayerNorm: one wave per row, vectorized, bf16 output
// ---------------------------------------------------------------------------
__global__ __launch_bounds__(256)
void layernorm_bf16(const float* __restrict__ x, const float* __restrict__ g,
                    const float* __restrict__ b, bf16_t* __restrict__ out) {
  int row  = blockIdx.x * 8 + (threadIdx.x >> 5);
  int lane = threadIdx.x & 31;
  const float*  xr = x + (size_t)row * CH;
  const float4* xv = (const float4*)xr;
  float s = 0.f, ss = 0.f;
  #pragma unroll
  for (int i = 0; i < 6; ++i) {              // 6*32*4 = 768
    float4 v = xv[lane + i*32];
    s  += v.x + v.y + v.z + v.w;
    ss += v.x*v.x + v.y*v.y + v.z*v.z + v.w*v.w;
  }
  #pragma unroll
  for (int m = 16; m >= 1; m >>= 1) { s += __shfl_xor(s, m, 32); ss += __shfl_xor(ss, m, 32); }
  float mean = s * (1.0f/CH);
  float var  = ss * (1.0f/CH) - mean*mean;
  float rstd = rsqrtf(var + 1e-5f);
  bf16_t* orow = out + (size_t)row * CH;
  #pragma unroll
  for (int i = 0; i < 6; ++i) {
    int c4 = (lane + i*32) * 4;
    float4 v  = xv[lane + i*32];
    float4 gg = *(const float4*)(g + c4);
    float4 bb = *(const float4*)(b + c4);
    v4bf o;
    o[0] = (__bf16)((v.x - mean) * rstd * gg.x + bb.x);
    o[1] = (__bf16)((v.y - mean) * rstd * gg.y + bb.y);
    o[2] = (__bf16)((v.z - mean) * rstd * gg.z + bb.z);
    o[3] = (__bf16)((v.w - mean) * rstd * gg.w + bb.w);
    *(v4bf*)(orow + c4) = o;
  }
}

// ---------------------------------------------------------------------------
// Tiled bf16 WMMA GEMM: C[M,N] = A[M,K] * BT[Npad,K]^T (+bias) (+epilogue)
// EPI 0: store f32 (bias optional)  EPI 1: residual add in place  EPI 2: GELU->bf16
// Block 128x128, 8 waves (4x2), wave tile 32x64, K-tile 64, double-buffered
// async global->LDS staging (ASYNCcnt pipelined).
// ---------------------------------------------------------------------------
#define LDK 72   // padded LDS row stride (elements): 144B, conflict-free frag reads

template <int EPI>
__global__ __launch_bounds__(256)
void gemm_bf16_wmma(const bf16_t* __restrict__ A, const bf16_t* __restrict__ BT,
                    const float* __restrict__ bias, float* __restrict__ outF,
                    bf16_t* __restrict__ outB, float* __restrict__ res,
                    int M, int N, int K) {
  __shared__ alignas(16) bf16_t As[2][128 * LDK];
  __shared__ alignas(16) bf16_t Bs[2][128 * LDK];
  const int tid  = threadIdx.x;
  const int lane = tid & 31;
  const int wave = tid >> 5;
  const int wm = wave >> 1;   // 0..3
  const int wn = wave & 1;    // 0..1
  const int bm = blockIdx.y * 128;
  const int bn = blockIdx.x * 128;
  const int l16 = lane & 15, lh = lane >> 4;

  v8f zero = {};
  v8f acc[2][4];
  #pragma unroll
  for (int i = 0; i < 2; ++i)
    #pragma unroll
    for (int j = 0; j < 4; ++j) acc[i][j] = zero;

  // stage one 128x64 A tile + 128x64 B tile: 8 x 16B per thread
  auto stage = [&](int buf, int k0) {
    #pragma unroll
    for (int it = 0; it < 4; ++it) {
      int chunk = tid + it * 256;        // 0..1023
      int row = chunk >> 3;              // 0..127
      int cc  = (chunk & 7) << 3;        // 0..56
      cp16_g2l(A  + (size_t)(bm + row) * K + k0 + cc, &As[buf][row * LDK + cc]);
      cp16_g2l(BT + (size_t)(bn + row) * K + k0 + cc, &Bs[buf][row * LDK + cc]);
    }
  };

  const int nstage = K >> 6;
  stage(0, 0);
  int buf = 0;
  for (int s = 0; s < nstage; ++s) {
    if (s + 1 < nstage) { stage(buf ^ 1, (s + 1) << 6); wait_async<8>(); }
    else                { wait_async<0>(); }
    __syncthreads();    // stage(s) visible to all waves

    #pragma unroll
    for (int ks = 0; ks < 2; ++ks) {
      v16bf af[2], bfr[4];
      #pragma unroll
      for (int mi = 0; mi < 2; ++mi) {
        int r = wm * 32 + mi * 16 + l16;
        af[mi] = ldsA16(&As[buf][r * LDK + ks * 32 + (lh << 3)]);
      }
      #pragma unroll
      for (int nj = 0; nj < 4; ++nj) {
        int r = wn * 64 + nj * 16 + l16;
        bfr[nj] = ldsB16(&Bs[buf][r * LDK + ks * 32 + (lh << 3)]);
      }
      #pragma unroll
      for (int mi = 0; mi < 2; ++mi)
        #pragma unroll
        for (int nj = 0; nj < 4; ++nj)
          acc[mi][nj] = wmma_bf16(af[mi], bfr[nj], acc[mi][nj]);
    }
    __syncthreads();    // all reads done before buffer is overwritten
    buf ^= 1;
  }

  // epilogue: C layout — VGPR e, lane: row = e + 8*(lane/16), col = lane%16
  #pragma unroll
  for (int mi = 0; mi < 2; ++mi) {
    #pragma unroll
    for (int nj = 0; nj < 4; ++nj) {
      #pragma unroll
      for (int e = 0; e < 8; ++e) {
        int row = bm + wm * 32 + mi * 16 + lh * 8 + e;
        int col = bn + wn * 64 + nj * 16 + l16;
        if (col < N) {
          float v = acc[mi][nj][e];
          if (bias) v += bias[col];
          size_t o = (size_t)row * N + col;
          if constexpr (EPI == 0) {
            outF[o] = v;
          } else if constexpr (EPI == 1) {
            res[o] += v;
          } else {
            float gl = 0.5f * v * (1.0f + erff(v * 0.70710678118f));
            outB[o] = (bf16_t)gl;
          }
        }
      }
    }
  }
}

// ---------------------------------------------------------------------------
// Flash attention helpers
// ---------------------------------------------------------------------------
template <bool MASK>
__device__ __forceinline__ void softmax_step(const v8f& s0, const v8f& s1,
                                             v8f (&o)[4], float (&rm)[8], float (&rl)[8],
                                             bf16_t* Pw, int qrow0, int kb,
                                             int l16, int lh) {
  #pragma unroll
  for (int e = 0; e < 8; ++e) {
    float v0 = s0[e], v1 = s1[e];
    if (MASK) {
      int q_t = qrow0 + lh * 8 + e;
      if (kb + l16 > q_t)      v0 = -1e30f;
      if (kb + 16 + l16 > q_t) v1 = -1e30f;
    }
    float mx = fmaxf(v0, v1);
    #pragma unroll
    for (int m = 8; m >= 1; m >>= 1) mx = fmaxf(mx, __shfl_xor(mx, m, 16));
    float mnew = fmaxf(rm[e], mx);
    float p0 = __expf(v0 - mnew);
    float p1 = __expf(v1 - mnew);
    float ps = p0 + p1;
    #pragma unroll
    for (int m = 8; m >= 1; m >>= 1) ps += __shfl_xor(ps, m, 16);
    float sc = __expf(rm[e] - mnew);
    rm[e] = mnew;
    rl[e] = rl[e] * sc + ps;
    #pragma unroll
    for (int j = 0; j < 4; ++j) o[j][e] *= sc;
    int prow = lh * 8 + e;
    Pw[prow * 40 + l16]      = (bf16_t)p0;
    Pw[prow * 40 + 16 + l16] = (bf16_t)p1;
  }
}

// ---------------------------------------------------------------------------
// Flash attention: 1 WG per (b, h, 64-query block); 4 waves x 16 query rows.
// K and V blocks staged cooperatively into LDS (bf16); qkv f32 [B*T, 3C];
// y bf16 [B*T, C].
// ---------------------------------------------------------------------------
__global__ __launch_bounds__(128)
void attention_kernel(const float* __restrict__ qkv, bf16_t* __restrict__ y) {
  __shared__ alignas(16) bf16_t Ks[32 * 72];        // [key][d]   row-major, padded
  __shared__ alignas(16) bf16_t Vs[HD * 40];        // [d][key]   transposed, padded
  __shared__ alignas(16) bf16_t Ps[4 * 16 * 40];    // per-wave P tiles
  const int tid = threadIdx.x, lane = tid & 31, w = tid >> 5;
  const int l16 = lane & 15, lh = lane >> 4;
  const int qb = blockIdx.x * 64;
  const int h = blockIdx.y, b = blockIdx.z;
  const size_t rstr = 3 * CH;
  const float* base = qkv + (size_t)b * SEQ * rstr;
  const int qrow0 = qb + 16 * w;

  // Q fragments (pre-scaled by 1/sqrt(64) = 0.125)
  const float* qp = base + (size_t)(qrow0 + l16) * rstr + h * HD;
  v16bf aq[2];
  #pragma unroll
  for (int f = 0; f < 2; ++f) {
    int kb8 = f * 32 + lh * 8;
    aq[f] = cat16(cvt8(qp + kb8, 0.125f), cvt8(qp + kb8 + 16, 0.125f));
  }

  v8f zero = {};
  v8f o[4];
  #pragma unroll
  for (int j = 0; j < 4; ++j) o[j] = zero;
  float rm[8], rl[8];
  #pragma unroll
  for (int e = 0; e < 8; ++e) { rm[e] = -1e30f; rl[e] = 0.f; }

  const int nkb = qb + 64;
  for (int kb = 0; kb < nkb; kb += 32) {
    __syncthreads();
    // stage K (row-major) and V (transposed) blocks: 32 keys x 64 d, float4 loads
    #pragma unroll
    for (int i = 0; i < 4; ++i) {
      int idx4 = tid + i * 128;        // 0..511 float4 chunks
      int kk = idx4 >> 4;              // 0..31
      int d4 = (idx4 & 15) << 2;       // 0..60
      const float* rowp = base + (size_t)(kb + kk) * rstr + h * HD + d4;
      float4 kv = *(const float4*)(rowp + CH);
      float4 vv = *(const float4*)(rowp + 2 * CH);
      *(v4bf*)(&Ks[kk * 72 + d4]) = cvt4(kv);
      Vs[(d4 + 0) * 40 + kk] = (__bf16)vv.x;
      Vs[(d4 + 1) * 40 + kk] = (__bf16)vv.y;
      Vs[(d4 + 2) * 40 + kk] = (__bf16)vv.z;
      Vs[(d4 + 3) * 40 + kk] = (__bf16)vv.w;
    }
    __syncthreads();

    if (kb > qrow0 + 15) continue;   // wave-uniform: this wave's rows fully masked

    // S = (Q*scale) K^T : two 16x16 tiles, K-dim 64; interleaved accumulation
    v8f s0 = zero, s1 = zero;
    {
      const bf16_t* kr0 = &Ks[l16 * 72 + (lh << 3)];
      const bf16_t* kr1 = &Ks[(16 + l16) * 72 + (lh << 3)];
      v16bf bk00 = ldsB16(kr0);
      v16bf bk10 = ldsB16(kr1);
      v16bf bk01 = ldsB16(kr0 + 32);
      v16bf bk11 = ldsB16(kr1 + 32);
      s0 = wmma_bf16(aq[0], bk00, s0);
      s1 = wmma_bf16(aq[0], bk10, s1);
      s0 = wmma_bf16(aq[1], bk01, s0);
      s1 = wmma_bf16(aq[1], bk11, s1);
    }

    // causal mask (only needed near the diagonal) + online softmax
    bf16_t* Pw = Ps + w * 16 * 40;
    if (kb + 31 <= qrow0)
      softmax_step<false>(s0, s1, o, rm, rl, Pw, qrow0, kb, l16, lh);
    else
      softmax_step<true>(s0, s1, o, rm, rl, Pw, qrow0, kb, l16, lh);
    asm volatile("s_wait_dscnt 0" ::: "memory");  // wave-local LDS RAW

    // O += P(16x32) * V(32x64)
    v16bf ap = ldsA16(&Pw[l16 * 40 + lh * 8]);
    #pragma unroll
    for (int j = 0; j < 4; ++j) {
      v16bf bv = ldsB16(&Vs[(j * 16 + l16) * 40 + lh * 8]);
      o[j] = wmma_bf16(ap, bv, o[j]);
    }
  }

  // normalize and store y (bf16)
  #pragma unroll
  for (int e = 0; e < 8; ++e) {
    float inv = 1.0f / rl[e];
    int q_t = qrow0 + lh * 8 + e;
    size_t orow = ((size_t)b * SEQ + q_t) * CH + h * HD;
    #pragma unroll
    for (int j = 0; j < 4; ++j)
      y[orow + j * 16 + l16] = (bf16_t)(o[j][e] * inv);
  }
}

// ---------------------------------------------------------------------------
extern "C" void kernel_launch(void* const* d_in, const int* in_sizes, int n_in,
                              void* d_out, int out_size, void* d_ws, size_t ws_size,
                              hipStream_t stream) {
  (void)in_sizes; (void)n_in; (void)out_size; (void)ws_size;
  const int*   idx    = (const int*)  d_in[0];
  const float* wte    = (const float*)d_in[1];
  const float* wpe    = (const float*)d_in[2];
  const float* ln1_g  = (const float*)d_in[3];
  const float* ln1_b  = (const float*)d_in[4];
  const float* w_attn = (const float*)d_in[5];
  const float* b_attn = (const float*)d_in[6];
  const float* w_ap   = (const float*)d_in[7];
  const float* b_ap   = (const float*)d_in[8];
  const float* ln2_g  = (const float*)d_in[9];
  const float* ln2_b  = (const float*)d_in[10];
  const float* w_fc   = (const float*)d_in[11];
  const float* b_fc   = (const float*)d_in[12];
  const float* w_mp   = (const float*)d_in[13];
  const float* b_mp   = (const float*)d_in[14];
  const float* lnf_g  = (const float*)d_in[15];
  const float* lnf_b  = (const float*)d_in[16];
  const float* w_head = (const float*)d_in[17];
  float* logits = (float*)d_out;

  const int VOCP = ((VOC + 127) / 128) * 128;   // 50304, zero-padded rows

  char* ws = (char*)d_ws;
  size_t off = 0;
  auto alloc = [&](size_t bytes) -> char* {
    char* p = ws + off;
    off += (bytes + 255) & ~(size_t)255;
    return p;
  };

  bf16_t* wTa = (bf16_t*)alloc((size_t)LAYERS * 3 * CH * CH * 2);
  bf16_t* wTp = (bf16_t*)alloc((size_t)LAYERS * CH * CH * 2);
  bf16_t* wTf = (bf16_t*)alloc((size_t)LAYERS * 4 * CH * CH * 2);
  bf16_t* wTm = (bf16_t*)alloc((size_t)LAYERS * 4 * CH * CH * 2);
  bf16_t* wTh = (bf16_t*)alloc((size_t)VOCP * CH * 2);
  float*  x   = (float*) alloc((size_t)MROWS * CH * 4);
  bf16_t* hbf = (bf16_t*)alloc((size_t)MROWS * CH * 2);
  float*  qkv = (float*) alloc((size_t)MROWS * 3 * CH * 4);
  bf16_t* ybf = (bf16_t*)alloc((size_t)MROWS * CH * 2);
  bf16_t* afc = (bf16_t*)alloc((size_t)MROWS * 4 * CH * 2);

  dim3 tb(32, 8);
  for (int l = 0; l < LAYERS; ++l) {
    transpose_cvt<<<dim3((3*CH)/32, CH/32),   tb, 0, stream>>>(w_attn + (size_t)l*CH*3*CH, wTa + (size_t)l*3*CH*CH, CH,   3*CH, 3*CH);
    transpose_cvt<<<dim3(CH/32,     CH/32),   tb, 0, stream>>>(w_ap   + (size_t)l*CH*CH,   wTp + (size_t)l*CH*CH,   CH,   CH,   CH);
    transpose_cvt<<<dim3((4*CH)/32, CH/32),   tb, 0, stream>>>(w_fc   + (size_t)l*CH*4*CH, wTf + (size_t)l*4*CH*CH, CH,   4*CH, 4*CH);
    transpose_cvt<<<dim3(CH/32,     (4*CH)/32), tb, 0, stream>>>(w_mp + (size_t)l*4*CH*CH, wTm + (size_t)l*4*CH*CH, 4*CH, CH,   CH);
  }
  transpose_cvt<<<dim3(VOCP/32, CH/32), tb, 0, stream>>>(w_head, wTh, CH, VOC, VOCP);

  embed_kernel<<<(MROWS*CH + 255)/256, 256, 0, stream>>>(idx, wte, wpe, x);

  for (int l = 0; l < LAYERS; ++l) {
    layernorm_bf16<<<MROWS/8, 256, 0, stream>>>(x, ln1_g + l*CH, ln1_b + l*CH, hbf);
    gemm_bf16_wmma<0><<<dim3((3*CH)/128, MROWS/128), 256, 0, stream>>>(
        hbf, wTa + (size_t)l*3*CH*CH, b_attn + l*3*CH, qkv, nullptr, nullptr, MROWS, 3*CH, CH);
    attention_kernel<<<dim3(SEQ/64, NH, BATCH), 128, 0, stream>>>(qkv, ybf);
    gemm_bf16_wmma<1><<<dim3(CH/128, MROWS/128), 256, 0, stream>>>(
        ybf, wTp + (size_t)l*CH*CH, b_ap + l*CH, nullptr, nullptr, x, MROWS, CH, CH);
    layernorm_bf16<<<MROWS/8, 256, 0, stream>>>(x, ln2_g + l*CH, ln2_b + l*CH, hbf);
    gemm_bf16_wmma<2><<<dim3((4*CH)/128, MROWS/128), 256, 0, stream>>>(
        hbf, wTf + (size_t)l*4*CH*CH, b_fc + l*4*CH, nullptr, afc, nullptr, MROWS, 4*CH, CH);
    gemm_bf16_wmma<1><<<dim3(CH/128, MROWS/128), 256, 0, stream>>>(
        afc, wTm + (size_t)l*4*CH*CH, b_mp + l*CH, nullptr, nullptr, x, MROWS, CH, 4*CH);
  }

  layernorm_bf16<<<MROWS/8, 256, 0, stream>>>(x, lnf_g, lnf_b, hbf);
  gemm_bf16_wmma<0><<<dim3(VOCP/128, MROWS/128), 256, 0, stream>>>(
      hbf, wTh, nullptr, logits, nullptr, nullptr, MROWS, VOC, CH);
}
